// NeuralFieldCosmo_66640712565063
// MI455X (gfx1250) — compile-verified, matrix operand order checked by reading, not run on hardware
//
#include <hip/hip_runtime.h>
#include <hip/hip_bf16.h>
#include <math.h>

// ---------------- problem constants (architecture of the MLP is fixed) ------
#define CIN   16
#define COUT  16
#define HDIM  32
#define EPSF  1e-5f
#define INV_RADIUS 1.0f

// packed-weights layout inside workspace (u32 units, after sums+counts)
#define W3PK     0      // 16 blocks * 32 lanes * 8 vgprs = 4096 u32 (bf16x2)
#define W2PK     4096   // 2 blocks  * 32 lanes * 8 vgprs = 512 u32
#define B3PK     4608   // 256 f32
#define SMALLPK  4864   // 288 f32: W1[96], b1, g1, beta1, b2, g2, beta2
#define WPK_TOTAL 5152  // u32

#define WPB 4           // waves per block (128 threads)
#define S2STRIDE 36     // layer-2 bounce row stride (f32): halves 32 banks apart
#define S3STRIDE 20     // layer-3 scratch row stride (f32): 16B-aligned rows
#define S3BUF    320    // one layer-3 scratch buffer (16 rows * 20)
#define SCR_F32  640    // per-wave scratch floats: max(16*36, 2*320)

typedef __attribute__((ext_vector_type(16))) __bf16 v16bf;
typedef __attribute__((ext_vector_type(8)))  float  v8f;

#if __has_builtin(__builtin_amdgcn_tanhf)
#define FAST_TANH(x) __builtin_amdgcn_tanhf(x)   // native V_TANH_F32 (TRANS)
#else
#define FAST_TANH(x) tanhf(x)
#endif

static __device__ inline unsigned pack_bf2(float a, float b) {
  union { __bf16 h[2]; unsigned u; } x;
  x.h[0] = (__bf16)a;      // low 16 bits  = even K
  x.h[1] = (__bf16)b;      // high 16 bits = odd K
  return x.u;
}

// load one 16-col B block (K=32 bf16) for this lane: 2x ds_load_b128
static __device__ inline v16bf ld_b16(const unsigned* p) {
  union { uint4 q[2]; v16bf v; } u;
  u.q[0] = *(const uint4*)(p);
  u.q[1] = *(const uint4*)(p + 4);
  return u.v;
}

// channel held by (vgpr v, sub j) for a lane in half `hw`, 16-bit A/B layout
static __device__ inline int kchan(int v, int j, int hw) {
  return ((v >> 2) << 4) + (hw << 3) + ((v & 3) << 1) + j;
}

// LDS ops from one wave execute in order (ISA ch.11), so cross-lane
// store->load visibility within a wave needs no counter drain — only a
// compiler/scheduler reordering barrier. Data deps on loaded values get
// their precise s_wait_dscnt from the compiler.
static __device__ inline void wave_lds_fence() {
  asm volatile("" ::: "memory");
  __builtin_amdgcn_wave_barrier();
}

// ---------------------------------------------------------------------------
__global__ void zero_ws_kernel(float* p, int n) {
  int i = blockIdx.x * blockDim.x + threadIdx.x;
  if (i < n) p[i] = 0.0f;
}

// Pre-pack W2/W3 into per-lane WMMA B-matrix bf16 layout; copy small consts.
__global__ void pack_weights_kernel(const float* __restrict__ W1, const float* __restrict__ b1,
                                    const float* __restrict__ g1, const float* __restrict__ be1,
                                    const float* __restrict__ W2, const float* __restrict__ b2,
                                    const float* __restrict__ g2, const float* __restrict__ be2,
                                    const float* __restrict__ W3, const float* __restrict__ b3,
                                    unsigned* __restrict__ wpk) {
  int tid = threadIdx.x;
  for (int idx = tid; idx < 4096; idx += blockDim.x) {        // W3: 32x256
    int o = idx >> 8, rem = idx & 255, lane = rem >> 3, v = rem & 7;
    int hw = lane >> 4, n = lane & 15;
    int k0 = kchan(v, 0, hw);
    int j  = (o << 4) | n;
    wpk[W3PK + idx] = pack_bf2(W3[k0 * 256 + j], W3[(k0 + 1) * 256 + j]);
  }
  for (int idx = tid; idx < 512; idx += blockDim.x) {         // W2: 32x32
    int o = idx >> 8, rem = idx & 255, lane = rem >> 3, v = rem & 7;
    int hw = lane >> 4, n = lane & 15;
    int k0 = kchan(v, 0, hw);
    int j  = (o << 4) | n;
    wpk[W2PK + idx] = pack_bf2(W2[k0 * 32 + j], W2[(k0 + 1) * 32 + j]);
  }
  float* f = (float*)wpk;
  for (int idx = tid; idx < 256; idx += blockDim.x) f[B3PK + idx] = b3[idx];
  for (int idx = tid; idx < 288; idx += blockDim.x) {
    float val;
    if      (idx <  96) val = W1[idx];
    else if (idx < 128) val = b1[idx - 96];
    else if (idx < 160) val = g1[idx - 128];
    else if (idx < 192) val = be1[idx - 160];
    else if (idx < 224) val = b2[idx - 192];
    else if (idx < 256) val = g2[idx - 224];
    else                val = be2[idx - 256];
    f[SMALLPK + idx] = val;
  }
}

// ---------------------------------------------------------------------------
// Fused neural-field + gather + per-edge matvec + scatter-add.
// One wave = 16 edges. EXEC stays all-ones around every WMMA.
__global__ __launch_bounds__(WPB * 32)
void nf_main_kernel(const int* __restrict__ in_edges, const int* __restrict__ out_edges,
                    const float* __restrict__ feat, const float* __restrict__ hood,
                    const unsigned* __restrict__ wpk,
                    float* __restrict__ sums, float* __restrict__ cnt, int E) {
  __shared__ __align__(16) unsigned sC[WPK_TOTAL];
  __shared__ __align__(16) float    sScrAll[WPB][SCR_F32];

  const int tid = threadIdx.x;
  {
    const uint4* src = (const uint4*)wpk;
    uint4* dst = (uint4*)sC;
    for (int i = tid; i < WPK_TOTAL / 4; i += WPB * 32) dst[i] = src[i];
  }
  __syncthreads();   // staging is split across waves -> block barrier once

  const unsigned* sW3 = sC;
  const unsigned* sW2 = sC + W2PK;
  const float*    sB3 = (const float*)(sC + B3PK);
  const float*    sSm = (const float*)(sC + SMALLPK);
  const float *sW1 = sSm, *sb1 = sSm + 96, *sg1 = sSm + 128, *se1 = sSm + 160,
              *sb2 = sSm + 192, *sg2 = sSm + 224, *se2 = sSm + 256;

  const int wave = tid >> 5;
  const int l    = tid & 31;
  const int hw   = l >> 4;     // which 16-lane half
  const int m    = l & 15;     // edge-in-tile (A layout) / column N (B,D layout)
  float* scr = sScrAll[wave];  // per-wave scratch, wave-local sync only

  const long gbase = ((long)blockIdx.x * WPB + wave) * 16;
  long eg  = gbase + m;
  bool alive = eg < E;
  long egc = alive ? eg : (long)(E - 1);

  const int ein  = in_edges[egc];   // this lane's own edge row (feature gather)
  const int eout = out_edges[egc];  // this lane's own edge scatter target
  const float x0 = hood[egc * 3 + 0] * INV_RADIUS;
  const float x1 = hood[egc * 3 + 1] * INV_RADIUS;
  const float x2 = hood[egc * 3 + 2] * INV_RADIUS;

  // ---- layer 1 (3->32) + LN + ReLU, built directly in WMMA A layout ----
  float h[16];
  float s = 0.f, sq = 0.f;
#pragma unroll
  for (int v = 0; v < 8; ++v)
#pragma unroll
    for (int j = 0; j < 2; ++j) {
      int c = kchan(v, j, hw);
      float t = sb1[c] + x0 * sW1[c] + x1 * sW1[32 + c] + x2 * sW1[64 + c];
      h[2 * v + j] = t; s += t; sq += t * t;
    }
  s  += __shfl_xor(s, 16, 32);
  sq += __shfl_xor(sq, 16, 32);
  float mean = s * (1.f / 32.f);
  float rstd = rsqrtf(sq * (1.f / 32.f) - mean * mean + EPSF);
  v16bf A1;
#pragma unroll
  for (int v = 0; v < 8; ++v)
#pragma unroll
    for (int j = 0; j < 2; ++j) {
      int c = kchan(v, j, hw);
      float y = (h[2 * v + j] - mean) * rstd * sg1[c] + se1[c];
      A1[2 * v + j] = (__bf16)fmaxf(y, 0.f);
    }

  // ---- layer 2 (32->32): two 16-col WMMA blocks, bounce D->LDS->A layout ----
  v8f cz = {};
#pragma unroll
  for (int o2 = 0; o2 < 2; ++o2) {
    v16bf B = ld_b16(sW2 + o2 * 256 + l * 8);
    v8f acc = __builtin_amdgcn_wmma_f32_16x16x32_bf16(false, A1, false, B,
                                                      (short)0, cz, false, false);
    int c2 = o2 * 16 + m;                 // output channel (D column)
    float bias = sb2[c2];
#pragma unroll
    for (int r = 0; r < 8; ++r)           // D row M = r + 8*hw
      scr[(r + (hw << 3)) * S2STRIDE + c2] = acc[r] + bias;
  }
  wave_lds_fence();

  // read back in A layout; LN2 + ReLU
  s = 0.f; sq = 0.f;
#pragma unroll
  for (int v = 0; v < 8; ++v)
#pragma unroll
    for (int j = 0; j < 2; ++j) {
      int c = kchan(v, j, hw);
      float t = scr[m * S2STRIDE + c];
      h[2 * v + j] = t; s += t; sq += t * t;
    }
  s  += __shfl_xor(s, 16, 32);
  sq += __shfl_xor(sq, 16, 32);
  mean = s * (1.f / 32.f);
  rstd = rsqrtf(sq * (1.f / 32.f) - mean * mean + EPSF);
  v16bf A2;
#pragma unroll
  for (int v = 0; v < 8; ++v)
#pragma unroll
    for (int j = 0; j < 2; ++j) {
      int c = kchan(v, j, hw);
      float y = (h[2 * v + j] - mean) * rstd * sg2[c] + se2[c];
      A2[2 * v + j] = (__bf16)fmaxf(y, 0.f);
    }
  wave_lds_fence();   // scr reused below with a different layout

  // this lane's feature sub-row: f[in_edges[e], 8*hw .. 8*hw+7]  (2x b128)
  const float4* frow = (const float4*)feat + ((unsigned)ein << 2);
  const float4 fva = frow[2 * hw + 0];
  const float4 fvb = frow[2 * hw + 1];
  const unsigned sumoff = ((unsigned)eout << 4);

  // ---- layer 3 (32->256) + tanh, fused per-edge matvec + scatter-add ----
  // For each 16-col block o:  D[M][N] = (x@W3)[edge M][i=N].
  // Transpose through wave-local, double-buffered LDS so the sum over i is
  // an in-lane dot; WAR across iterations is handled by the buffer flip.
#pragma unroll 4
  for (int o = 0; o < 16; ++o) {
    float* buf = scr + (o & 1) * S3BUF;
    v16bf B = ld_b16(sW3 + o * 256 + l * 8);
    v8f acc = __builtin_amdgcn_wmma_f32_16x16x32_bf16(false, A2, false, B,
                                                      (short)0, cz, false, false);
    float bias = sB3[o * 16 + m];
#pragma unroll
    for (int r = 0; r < 8; ++r)            // write w[e=M][i=m], M = r + 8*hw
      buf[(r + (hw << 3)) * S3STRIDE + m] = FAST_TANH(acc[r] + bias);
    wave_lds_fence();

    // lane reads its OWN edge's row: i = 8*hw .. 8*hw+7 (2x ds_load_b128)
    const float4 q0 = *(const float4*)&buf[m * S3STRIDE + 8 * hw + 0];
    const float4 q1 = *(const float4*)&buf[m * S3STRIDE + 8 * hw + 4];
    float part = q0.x * fva.x + q0.y * fva.y + q0.z * fva.z + q0.w * fva.w
               + q1.x * fvb.x + q1.y * fvb.y + q1.z * fvb.z + q1.w * fvb.w;
    part += __shfl_xor(part, 16, 32);      // combine the two halves
    if (hw == 0 && alive)
      atomicAdd(&sums[sumoff + (unsigned)o], part);
  }

  if (hw == 0 && alive) atomicAdd(&cnt[eout], 1.0f);
}

// ---------------------------------------------------------------------------
__global__ void finalize_kernel(const float* __restrict__ sums, const float* __restrict__ cnt,
                                float* __restrict__ out, int n) {
  int i = blockIdx.x * blockDim.x + threadIdx.x;
  if (i < n) out[i] = sums[i] / fmaxf(cnt[i >> 4], 1.0f);
}

// ---------------------------------------------------------------------------
extern "C" void kernel_launch(void* const* d_in, const int* in_sizes, int n_in,
                              void* d_out, int out_size, void* d_ws, size_t ws_size,
                              hipStream_t stream) {
  const int*   in_e  = (const int*)d_in[0];
  const int*   out_e = (const int*)d_in[1];
  const float* feat  = (const float*)d_in[2];
  const float* hood  = (const float*)d_in[3];
  const float* W1  = (const float*)d_in[4];
  const float* b1  = (const float*)d_in[5];
  const float* g1  = (const float*)d_in[6];
  const float* be1 = (const float*)d_in[7];
  const float* W2  = (const float*)d_in[8];
  const float* b2  = (const float*)d_in[9];
  const float* g2  = (const float*)d_in[10];
  const float* be2 = (const float*)d_in[11];
  const float* W3  = (const float*)d_in[12];
  const float* b3  = (const float*)d_in[13];

  const int E = in_sizes[0];
  const int N = in_sizes[2] / CIN;

  float* ws_f = (float*)d_ws;
  float* sums = ws_f;                       // N*COUT f32
  float* cnt  = ws_f + (size_t)N * COUT;    // N f32
  size_t wpk_off = (((size_t)N * (COUT + 1) + 3) / 4) * 4;  // 16B aligned
  unsigned* wpk = (unsigned*)d_ws + wpk_off;

  int zn = N * (COUT + 1);
  zero_ws_kernel<<<(zn + 255) / 256, 256, 0, stream>>>(ws_f, zn);
  pack_weights_kernel<<<1, 256, 0, stream>>>(W1, b1, g1, be1, W2, b2, g2, be2, W3, b3, wpk);

  long ntiles  = ((long)E + 15) / 16;
  long nblocks = (ntiles + WPB - 1) / WPB;
  nf_main_kernel<<<dim3((unsigned)nblocks), WPB * 32, 0, stream>>>(
      in_e, out_e, feat, hood, wpk, sums, cnt, E);

  int on = N * COUT;
  finalize_kernel<<<(on + 255) / 256, 256, 0, stream>>>(sums, cnt, (float*)d_out, on);
}
